// TransformerCore_72121090834714
// MI455X (gfx1250) — matrix operand order repaired and checked by reading, not compiled, not run on hardware
//
#include <hip/hip_runtime.h>
#include <hip/hip_bf16.h>

typedef __attribute__((ext_vector_type(16))) _Float16 v16h;
typedef __attribute__((ext_vector_type(8)))  float    v8f;
typedef __attribute__((ext_vector_type(4)))  unsigned int v4u;
typedef __attribute__((ext_vector_type(8)))  int      v8i_t;
typedef __attribute__((ext_vector_type(4)))  int      v4i_t;

#define N_AG  5
#define EMB   64
#define T_STEPS 128
#define B_EP  128
#define IN_SZ 256

// packed f16 weight offsets inside d_ws (units: halfs)
#define OFF_WENC   0u        // 256x64
#define OFF_ATTNW  16384u    // 64x192
#define OFF_ATTNPW 28672u    // 64x64
#define OFF_QW     32768u    // 64x64
#define OFF_KVW    36864u    // 64x128
#define OFF_CPW    45056u    // 64x64
#define OFF_FCW    49152u    // 64x256
#define OFF_MPW    65536u    // 256x64
#define OFF_WMEM   81920u    // 64x64
#define OFF_WDEC   86016u    // 64x256

// ---------------- weight pre-pack: f32 row-major (K x N) -> f16 B-fragment order
// b[i] of lane l for tile (kt,nt) = W[kt*32 + (l>>4)*8 + (i&7) + (i>>3)*16][nt*16 + (l&15)]
// stored at dst[(((nt*KT + kt)*32 + l)*16 + i)]
__global__ void pack_weight_kernel(const float* __restrict__ src, _Float16* __restrict__ dst,
                                   int K, int N) {
  int KT = K >> 5;
  int total = K * N;
  for (int idx = blockIdx.x * blockDim.x + threadIdx.x; idx < total;
       idx += gridDim.x * blockDim.x) {
    int i    = idx & 15;
    int lane = (idx >> 4) & 31;
    int rest = idx >> 9;
    int kt   = rest % KT;
    int nt   = rest / KT;
    int col  = nt * 16 + (lane & 15);
    int krow = kt * 32 + ((lane >> 4) << 3) + (i & 7) + ((i >> 3) << 4);
    dst[idx] = (_Float16)src[krow * N + col];
  }
}

// ---------------- TDM: async DMA of one obs tile (5 x 256 f32) into LDS -----
// D# per CDNA5 ISA ch.8: group0 {count=1, lds_addr, global_addr, type=2},
// group1 {data_size=4B, tensor 256x5, tile 256x5, dim0_stride=256}, g2/g3 = 0.
__device__ __forceinline__ void tdm_load_obs(unsigned lds_addr, const float* gptr) {
  unsigned long long ga = (unsigned long long)(size_t)gptr;
  v4u g0;
  g0[0] = 1u;                                              // count=1 (valid), no gather
  g0[1] = lds_addr;                                        // LDS byte address
  g0[2] = (unsigned)(ga & 0xFFFFFFFFu);                    // global_addr[31:0]
  g0[3] = (unsigned)((ga >> 32) & 0x01FFFFFFu) | (2u << 30); // addr[56:32] | type=2
  v8i_t g1;
  g1[0] = (2 << 16);            // workgroup_mask=0, data_size=2 (4 bytes)
  g1[1] = (int)(256u << 16);    // tensor_dim0[15:0]=256 @bits 63:48
  g1[2] = (int)(5u << 16);      // tensor_dim0[31:16]=0, tensor_dim1[15:0]=5
  g1[3] = (int)(256u << 16);    // tensor_dim1[31:16]=0, tile_dim0=256
  g1[4] = 5;                    // tile_dim1=5, tile_dim2=0
  g1[5] = 256;                  // tensor_dim0_stride[31:0]=256
  g1[6] = 0;                    // stride hi / dim1_stride lo
  g1[7] = 0;
  v4i_t z4 = {0, 0, 0, 0};
#if __clang_major__ >= 23
  v8i_t z8 = {0, 0, 0, 0, 0, 0, 0, 0};
  __builtin_amdgcn_tensor_load_to_lds(g0, g1, z4, z4, z8, 0);
#else
  __builtin_amdgcn_tensor_load_to_lds(g0, g1, z4, z4, 0);
#endif
}

// ---------------- WMMA fragment loaders (wave32, 16x16x32 f16) ----------------
__device__ __forceinline__ v16h load_A_frag(const _Float16* A, int strideH,
                                            int m0, int k0, int lane) {
  int row = m0 + (lane & 15);
  int kb  = k0 + ((lane >> 4) << 3);
  const _Float16* p = A + row * strideH + kb;
  v16h a;
#pragma unroll
  for (int i = 0; i < 8; ++i) { a[i] = p[i]; a[8 + i] = p[16 + i]; }
  return a;
}

__device__ __forceinline__ v16h load_B_frag(const _Float16* Wp, int KT,
                                            int kt, int nt, int lane) {
  const _Float16* p = Wp + (((size_t)(nt * KT + kt) * 32 + lane) << 4);
  v16h b;
#pragma unroll
  for (int i = 0; i < 16; ++i) b[i] = p[i];
  return b;
}

// C = A(64xK, LDS f16) * Wpacked(KxN, global f16) + bias ; f32 out to LDS (outStride)
__device__ __forceinline__ void gemm_tiles(const _Float16* A, int As,
                                           const _Float16* Wp,
                                           float* outLds, int outStride,
                                           const float* bias,
                                           int K, int N, int Mtiles, int tid) {
  int lane = tid & 31, wave = tid >> 5;
  int KT = K >> 5, NT = N >> 4;
  int total = Mtiles * NT;
  for (int tile = wave; tile < total; tile += 4) {
    int mt = tile / NT, nt = tile % NT;
    v8f c = {};
    for (int kt = 0; kt < KT; ++kt) {
      v16h a = load_A_frag(A, As, mt * 16, kt * 32, lane);
      v16h b = load_B_frag(Wp, KT, kt, nt, lane);
      c = __builtin_amdgcn_wmma_f32_16x16x32_f16(false, a, false, b,
                                                 (short)0, c, false, false);
    }
    int col   = nt * 16 + (lane & 15);
    int rbase = mt * 16 + ((lane >> 4) << 3);
    float bv = bias ? bias[col] : 0.f;
#pragma unroll
    for (int r = 0; r < 8; ++r)
      outLds[(rbase + r) * outStride + col] = c[r] + bv;
  }
}

// dec GEMM: core(16x64 f16, rows<5 valid) @ W_dec(64x256) + b_dec -> global f32
__device__ __forceinline__ void gemm_dec(const _Float16* A, const _Float16* Wp,
                                         const float* bias, float* out, int tid) {
  int lane = tid & 31, wave = tid >> 5;
  for (int nt = wave; nt < 16; nt += 4) {
    v8f c = {};
    for (int kt = 0; kt < 2; ++kt) {
      v16h a = load_A_frag(A, 64, 0, kt * 32, lane);
      v16h b = load_B_frag(Wp, 2, kt, nt, lane);
      c = __builtin_amdgcn_wmma_f32_16x16x32_f16(false, a, false, b,
                                                 (short)0, c, false, false);
    }
    int col   = nt * 16 + (lane & 15);
    int rbase = (lane >> 4) << 3;
#pragma unroll
    for (int r = 0; r < 8; ++r) {
      int row = rbase + r;
      if (row < N_AG) out[row * IN_SZ + col] = c[r] + bias[col];
    }
  }
}

// ---------------- VALU helpers ----------------
__device__ __forceinline__ void ln_rows_f16(const float* h, _Float16* dst,
                                            const float* g, const float* bb,
                                            int rows, int tid) {
  if (tid < 64) {
    int r = tid;
    if (r < rows) {
      float m = 0.f;
      for (int e = 0; e < 64; ++e) m += h[r * 64 + e];
      m *= (1.f / 64.f);
      float v = 0.f;
      for (int e = 0; e < 64; ++e) { float d = h[r * 64 + e] - m; v += d * d; }
      float inv = rsqrtf(v * (1.f / 64.f) + 1e-5f);
      for (int e = 0; e < 64; ++e)
        dst[r * 64 + e] = (_Float16)((h[r * 64 + e] - m) * inv * g[e] + bb[e]);
    } else {
      for (int e = 0; e < 64; ++e) dst[r * 64 + e] = (_Float16)0.f;
    }
  }
}

__device__ __forceinline__ void ln_rows_f32(const float* h, float* dst,
                                            const float* g, const float* bb,
                                            int rows, int tid) {
  if (tid < 64) {
    int r = tid;
    if (r < rows) {
      float m = 0.f;
      for (int e = 0; e < 64; ++e) m += h[r * 64 + e];
      m *= (1.f / 64.f);
      float v = 0.f;
      for (int e = 0; e < 64; ++e) { float d = h[r * 64 + e] - m; v += d * d; }
      float inv = rsqrtf(v * (1.f / 64.f) + 1e-5f);
      for (int e = 0; e < 64; ++e)
        dst[r * 64 + e] = (h[r * 64 + e] - m) * inv * g[e] + bb[e];
    } else {
      for (int e = 0; e < 64; ++e) dst[r * 64 + e] = 0.f;
    }
  }
}

// attention in fp32 VALU. qkv row stride 192: q@0, k@64, v@128.
__device__ __forceinline__ void attn_valu(const float* qkv, float* obuf,
                                          int L, int Lk, bool causal,
                                          bool shared_kv, int tid) {
  int total = N_AG * 4 * L;
  for (int w = tid; w < total; w += 128) {
    int i  = w % L;
    int ah = w / L;
    int a  = ah >> 2, hd = ah & 3;
    const float* qp = qkv + (a * L + i) * 192 + hd * 16;
    float s[10];
    float mx = -1e30f;
    int jmax = causal ? (i + 1) : Lk;
    for (int j = 0; j < jmax; ++j) {
      const float* kp = qkv + (shared_kv ? j : (a * L + j)) * 192 + 64 + hd * 16;
      float d = 0.f;
#pragma unroll
      for (int e = 0; e < 16; ++e) d += qp[e] * kp[e];
      s[j] = d * 0.25f;           // 1/sqrt(16)
      mx = fmaxf(mx, s[j]);
    }
    float sum = 0.f;
    for (int j = 0; j < jmax; ++j) { s[j] = expf(s[j] - mx); sum += s[j]; }
    float inv = 1.f / sum;
    float o[16];
#pragma unroll
    for (int e = 0; e < 16; ++e) o[e] = 0.f;
    for (int j = 0; j < jmax; ++j) {
      const float* vp = qkv + (shared_kv ? j : (a * L + j)) * 192 + 128 + hd * 16;
      float aj = s[j] * inv;
#pragma unroll
      for (int e = 0; e < 16; ++e) o[e] += aj * vp[e];
    }
    float* op = obuf + (a * L + i) * 64 + hd * 16;
#pragma unroll
    for (int e = 0; e < 16; ++e) op[e] = o[e];
  }
}

__device__ __forceinline__ void cvt_f16_pad(const float* src, _Float16* dst,
                                            int rows, int tid) {
  for (int i = tid; i < 64 * 64; i += 128)
    dst[i] = ((i >> 6) < rows) ? (_Float16)src[i] : (_Float16)0.f;
}

struct Params {
  const float *obs, *b_enc, *wpe;
  const float *ln1_g, *ln1_b, *attn_b, *attn_pb;
  const float *lnc_g, *lnc_b, *q_b, *kv_b, *c_pb;
  const float *ln2_g, *ln2_b, *fc_b, *m_pb;
  const float *lnf_g, *lnf_b, *b_dec;
  const _Float16 *pWenc, *pAttnW, *pAttnPW, *pQW, *pKVW, *pCPW, *pFCW, *pMPW, *pWmem, *pWdec;
  float* out;
};

// LDS: h 16K | buf 64K | obuf 16K | hist 10K | mem 1.25K | ci 1.25K
//      | hn16 8K | m16 32K | sA16 2K | obs_stage 5K  => 159232 bytes
#define SMEM_BYTES 159232

__global__ void __launch_bounds__(128)
recurrent_scan_kernel(Params P) {
  extern __shared__ unsigned char smem[];
  float* h    = (float*)smem;                   // 64x64
  float* buf  = h + 64 * 64;                    // 64x256 (multi-use, stride varies)
  float* obuf = buf + 64 * 256;                 // 64x64
  float* hist = obuf + 64 * 64;                 // 5x8x64
  float* memv = hist + N_AG * 8 * 64;           // 5x64
  float* ci   = memv + N_AG * 64;               // 5x64
  _Float16* hn16 = (_Float16*)(ci + N_AG * 64); // 64x64
  _Float16* m16  = hn16 + 64 * 64;              // 64x256
  _Float16* sA16 = m16 + 64 * 256;              // 16x64
  float* stage   = (float*)(sA16 + 16 * 64);    // 5x256 obs staging (TDM target)

  int tid = threadIdx.x;
  int b   = blockIdx.x;

  for (int i = tid; i < N_AG * 8 * 64; i += 128) hist[i] = 0.f;
  for (int i = tid; i < N_AG * 64; i += 128) memv[i] = 0.f;
  __syncthreads();

  // TDM pipeline prologue: kick off DMA of the first obs tile (wave 0 issues).
  unsigned stage_lds = (unsigned)(size_t)(void*)stage;   // LDS aperture keeps offset in [31:0]
  if (tid < 32)
    tdm_load_obs(stage_lds, P.obs + ((size_t)b * T_STEPS) * N_AG * IN_SZ);

  for (int t = 0; t < T_STEPS; ++t) {
    int L = (t == 0) ? 9 : 10;
    int rows = N_AG * L;
    bool cross = (t > 0);

    // ---- consume staged obs tile, convert to f16 A-fragments in m16
    if (tid < 32) __builtin_amdgcn_s_wait_tensorcnt(0);
    __syncthreads();
    for (int i = tid; i < 16 * IN_SZ; i += 128) {
      int r = i >> 8, c = i & 255;
      m16[i] = (r < N_AG) ? (_Float16)stage[r * IN_SZ + c] : (_Float16)0.f;
    }
    __syncthreads();
    // overlap next step's obs DMA with the whole step's compute
    if (t + 1 < T_STEPS && tid < 32)
      tdm_load_obs(stage_lds, P.obs + ((size_t)b * T_STEPS + t + 1) * N_AG * IN_SZ);

    // ---- encoder: ci = obs[b,t] @ W_enc + b_enc
    gemm_tiles(m16, IN_SZ, P.pWenc, buf, 64, P.b_enc, IN_SZ, 64, 1, tid);
    __syncthreads();
    for (int i = tid; i < N_AG * 64; i += 128) ci[i] = buf[i];
    __syncthreads();

    // ---- build input rows: h[r] = {mem|hist|ci} + wpe[j]
    for (int i = tid; i < 64 * 64; i += 128) {
      int r = i >> 6, e = i & 63;
      float val = 0.f;
      if (r < rows) {
        int a = r / L, j = r % L;
        float base;
        if (t == 0) base = (j < 8) ? hist[(a * 8 + j) * 64 + e] : ci[a * 64 + e];
        else        base = (j == 0) ? memv[a * 64 + e]
                         : (j < 9)  ? hist[(a * 8 + (j - 1)) * 64 + e]
                                    : ci[a * 64 + e];
        val = base + P.wpe[j * 64 + e];
      }
      h[i] = val;
    }
    __syncthreads();

    // ---- self attention
    ln_rows_f16(h, hn16, P.ln1_g, P.ln1_b, rows, tid);
    __syncthreads();
    gemm_tiles(hn16, 64, P.pAttnW, buf, 192, P.attn_b, 64, 192, 4, tid); // qkv
    __syncthreads();
    attn_valu(buf, obuf, L, L, true, false, tid);
    __syncthreads();
    cvt_f16_pad(obuf, hn16, rows, tid);
    __syncthreads();
    gemm_tiles(hn16, 64, P.pAttnPW, obuf, 64, P.attn_pb, 64, 64, 4, tid);
    __syncthreads();
    for (int i = tid; i < rows * 64; i += 128) h[i] += obuf[i];
    __syncthreads();

    // ---- cross attention (t >= 1): keys/values from mem (shared across agents)
    if (cross) {
      ln_rows_f16(h, hn16, P.lnc_g, P.lnc_b, rows, tid);
      for (int i = tid; i < 16 * 64; i += 128)
        sA16[i] = (i < N_AG * 64) ? (_Float16)memv[i] : (_Float16)0.f;
      __syncthreads();
      gemm_tiles(hn16, 64, P.pQW, buf, 192, P.q_b, 64, 64, 4, tid);         // q -> cols 0..63
      gemm_tiles(sA16, 64, P.pKVW, buf + 64, 192, P.kv_b, 64, 128, 1, tid); // k,v -> cols 64..191
      __syncthreads();
      attn_valu(buf, obuf, L, N_AG, false, true, tid);
      __syncthreads();
      cvt_f16_pad(obuf, hn16, rows, tid);
      __syncthreads();
      gemm_tiles(hn16, 64, P.pCPW, obuf, 64, P.c_pb, 64, 64, 4, tid);
      __syncthreads();
      for (int i = tid; i < rows * 64; i += 128) h[i] += obuf[i];
      __syncthreads();
    }

    // ---- MLP
    ln_rows_f16(h, hn16, P.ln2_g, P.ln2_b, rows, tid);
    __syncthreads();
    gemm_tiles(hn16, 64, P.pFCW, buf, 256, P.fc_b, 64, 256, 4, tid);
    __syncthreads();
    for (int i = tid; i < 64 * 256; i += 128) {     // GELU(tanh) + f16 convert
      float x = ((i >> 8) < rows) ? buf[i] : 0.f;
      float g = 0.5f * x * (1.f + tanhf(0.7978845608f * (x + 0.044715f * x * x * x)));
      m16[i] = (_Float16)g;
    }
    __syncthreads();
    gemm_tiles(m16, 256, P.pMPW, obuf, 64, P.m_pb, 256, 64, 4, tid);
    __syncthreads();
    for (int i = tid; i < rows * 64; i += 128) h[i] += obuf[i];
    __syncthreads();

    // ---- final LN, mem / core extraction, decode
    ln_rows_f32(h, obuf, P.lnf_g, P.lnf_b, rows, tid);  // x -> obuf (f32)
    __syncthreads();
    for (int i = tid; i < 16 * 64; i += 128) {
      int r = i >> 6, e = i & 63;
      _Float16 vm = (_Float16)0.f, vc = (_Float16)0.f;
      if (r < N_AG) {
        int msrc = (t == 0) ? (r * L + (L - 1)) : (r * L);  // t=0: mem from core row
        vm = (_Float16)obuf[msrc * 64 + e];
        vc = (_Float16)obuf[(r * L + (L - 1)) * 64 + e];    // core = last row
      }
      sA16[i] = vm;
      hn16[i] = vc;
    }
    __syncthreads();
    gemm_tiles(sA16, 64, P.pWmem, buf, 64, nullptr, 64, 64, 1, tid);  // new mem
    gemm_dec(hn16, P.pWdec, P.b_dec,
             P.out + (((size_t)b * T_STEPS + t) * N_AG) * IN_SZ, tid);
    __syncthreads();

    // ---- state update: mem, hist (shift by 1, append ci)
    for (int i = tid; i < N_AG * 64; i += 128) memv[i] = buf[i];
    float* tmpb = buf + 4096;  // free region of buf as shift temp
    for (int i = tid; i < N_AG * 7 * 64; i += 128) {
      int a = i / (7 * 64), j = (i % (7 * 64)) >> 6, e = i & 63;
      tmpb[i] = hist[(a * 8 + j + 1) * 64 + e];
    }
    __syncthreads();
    for (int i = tid; i < N_AG * 7 * 64; i += 128) {
      int a = i / (7 * 64), j = (i % (7 * 64)) >> 6, e = i & 63;
      hist[(a * 8 + j) * 64 + e] = tmpb[i];
    }
    for (int i = tid; i < N_AG * 64; i += 128)
      hist[((i >> 6) * 8 + 7) * 64 + (i & 63)] = ci[i];
    __syncthreads();
  }
}

extern "C" void kernel_launch(void* const* d_in, const int* in_sizes, int n_in,
                              void* d_out, int out_size, void* d_ws, size_t ws_size,
                              hipStream_t stream) {
  (void)in_sizes; (void)n_in; (void)out_size; (void)ws_size;
  _Float16* ws = (_Float16*)d_ws;

  // pack all weight matrices to f16 B-fragment order
  pack_weight_kernel<<<64, 256, 0, stream>>>((const float*)d_in[1],  ws + OFF_WENC,   256, 64);
  pack_weight_kernel<<<64, 256, 0, stream>>>((const float*)d_in[6],  ws + OFF_ATTNW,  64, 192);
  pack_weight_kernel<<<64, 256, 0, stream>>>((const float*)d_in[8],  ws + OFF_ATTNPW, 64, 64);
  pack_weight_kernel<<<64, 256, 0, stream>>>((const float*)d_in[12], ws + OFF_QW,     64, 64);
  pack_weight_kernel<<<64, 256, 0, stream>>>((const float*)d_in[14], ws + OFF_KVW,    64, 128);
  pack_weight_kernel<<<64, 256, 0, stream>>>((const float*)d_in[16], ws + OFF_CPW,    64, 64);
  pack_weight_kernel<<<64, 256, 0, stream>>>((const float*)d_in[20], ws + OFF_FCW,    64, 256);
  pack_weight_kernel<<<64, 256, 0, stream>>>((const float*)d_in[22], ws + OFF_MPW,    256, 64);
  pack_weight_kernel<<<64, 256, 0, stream>>>((const float*)d_in[26], ws + OFF_WMEM,   64, 64);
  pack_weight_kernel<<<64, 256, 0, stream>>>((const float*)d_in[27], ws + OFF_WDEC,   64, 256);

  Params P;
  P.obs    = (const float*)d_in[0];
  P.b_enc  = (const float*)d_in[2];
  P.wpe    = (const float*)d_in[3];
  P.ln1_g  = (const float*)d_in[4];  P.ln1_b = (const float*)d_in[5];
  P.attn_b = (const float*)d_in[7];  P.attn_pb = (const float*)d_in[9];
  P.lnc_g  = (const float*)d_in[10]; P.lnc_b = (const float*)d_in[11];
  P.q_b    = (const float*)d_in[13]; P.kv_b  = (const float*)d_in[15];
  P.c_pb   = (const float*)d_in[17];
  P.ln2_g  = (const float*)d_in[18]; P.ln2_b = (const float*)d_in[19];
  P.fc_b   = (const float*)d_in[21]; P.m_pb  = (const float*)d_in[23];
  P.lnf_g  = (const float*)d_in[24]; P.lnf_b = (const float*)d_in[25];
  P.b_dec  = (const float*)d_in[28];
  P.pWenc  = ws + OFF_WENC;   P.pAttnW  = ws + OFF_ATTNW;
  P.pAttnPW= ws + OFF_ATTNPW; P.pQW     = ws + OFF_QW;
  P.pKVW   = ws + OFF_KVW;    P.pCPW    = ws + OFF_CPW;
  P.pFCW   = ws + OFF_FCW;    P.pMPW    = ws + OFF_MPW;
  P.pWmem  = ws + OFF_WMEM;   P.pWdec   = ws + OFF_WDEC;
  P.out    = (float*)d_out;

  (void)hipFuncSetAttribute(reinterpret_cast<const void*>(recurrent_scan_kernel),
                            hipFuncAttributeMaxDynamicSharedMemorySize, SMEM_BYTES);
  recurrent_scan_kernel<<<B_EP, 128, SMEM_BYTES, stream>>>(P);
}
